// Sequence_24781961298178
// MI455X (gfx1250) — matrix operand order repaired and checked by reading, not compiled, not run on hardware
//
#include <hip/hip_runtime.h>
#include <hip/hip_bf16.h>

// ---------------------------------------------------------------------------
// 2-layer LSTM scheduled-sampling scan, T=1024, B=512, H=64.
// Batch rows are independent -> 32 blocks x 16 rows, whole T-loop in-kernel.
// Layer-1 matvec batch [16x64]x[64x256] via v_wmma_f32_16x16x32_f16 (f32 acc).
// Sigmoid/tanh use v_exp_f32 + v_rcp_f32 (no IEEE divide sequences).
// ---------------------------------------------------------------------------

typedef __attribute__((ext_vector_type(16))) _Float16 v16h;
typedef __attribute__((ext_vector_type(4)))  _Float16 v4h;
typedef __attribute__((ext_vector_type(8)))  float    v8f;
typedef __attribute__((ext_vector_type(4)))  float    v4f;

#define T_STEPS 1024
#define BSZ     512
#define H1      64
#define ROWS    16     // batch rows per block
#define H1PAD   68     // f16 LDS row stride (8B-aligned rows, bank-spread)
#define ZPAD    20     // f32 LDS row stride for gate exchange (16B-aligned)

// v_rcp_f32 instead of IEEE-correct division (saves ~8 VALU per divide)
__device__ __forceinline__ float rcpf(float x)      { return __builtin_amdgcn_rcpf(x); }
__device__ __forceinline__ float sigf(float x)      { return rcpf(1.0f + __expf(-x)); }
__device__ __forceinline__ float tanh_fast(float x) {
    return fmaf(-2.0f, rcpf(__expf(2.0f * x) + 1.0f), 1.0f);
}

union F16V { v16h v; v4h q[4]; _Float16 h[16]; };

__global__ __launch_bounds__(256, 1)
void lstm2_scan_kernel(const float* __restrict__ seq,
                       const unsigned char* __restrict__ use_gt,   // jax bool_: 1 byte
                       const float* __restrict__ W_ih1, const float* __restrict__ W_hh1,
                       const float* __restrict__ b_ih1, const float* __restrict__ b_hh1,
                       const float* __restrict__ W_ih2, const float* __restrict__ W_hh2,
                       const float* __restrict__ b_ih2, const float* __restrict__ b_hh2,
                       float* __restrict__ out)
{
    __shared__ __align__(16) _Float16 h1s[ROWS * H1PAD];   // h1 state (f16) [row][k]
    __shared__ __align__(16) float    zT[256 * ZPAD];      // z exchange [z-col 0..255][row]
    __shared__ __align__(16) float    xbuf[ROWS];          // scheduled-sampling input
    __shared__ __align__(16) float    h2buf[ROWS];         // h2 state
    __shared__ __align__(16) float    z2buf[ROWS * 4];     // layer-2 gate exchange

    const int tid    = threadIdx.x;
    const int lane   = tid & 31;
    const int wave   = tid >> 5;
    const int hi     = lane >> 4;      // lane half: C rows v+8*hi, B K-half
    const int ln     = lane & 15;
    const int gate   = wave >> 1;      // 0..3 = i,f,g,o
    const int npair  = wave & 1;       // which pair of 16-col tiles of this gate
    const int batch0 = blockIdx.x * ROWS;

    // ---- one-time: W_hh1 B-fragments (f16) into registers -----------------
    // B (32x16 KxN) layout: lane holds col n=ln; element i -> K = (i<8?0:16) + hi*8 + (i&7).
    F16V  Bw[2][2];                    // [n-tile in wave][k-chunk]
    float wih1c[2], b1c[2];
    for (int j = 0; j < 2; ++j) {
        const int ncol = gate * 64 + (npair * 2 + j) * 16 + ln;   // z column 0..255
        for (int kc = 0; kc < 2; ++kc)
            #pragma unroll
            for (int i = 0; i < 16; ++i)
                Bw[j][kc].h[i] = (_Float16)W_hh1[ncol * H1 + kc * 32 + hi * 8 + (i & 8) + i];
        wih1c[j] = W_ih1[ncol];
        b1c[j]   = b_ih1[ncol] + b_hh1[ncol];
    }

    // ---- layer-2 constants (used by wave 0; cols 4..15 padded with zeros) --
    F16V  B2[2];
    float whh2c = 0.0f, b2c = 0.0f;
    for (int kc = 0; kc < 2; ++kc)
        #pragma unroll
        for (int i = 0; i < 16; ++i)
            B2[kc].h[i] = (ln < 4) ? (_Float16)W_ih2[ln * H1 + kc * 32 + hi * 8 + (i & 8) + i]
                                   : (_Float16)0.0f;
    if (ln < 4) { whh2c = W_hh2[ln]; b2c = b_ih2[ln] + b_hh2[ln]; }

    // ---- init state --------------------------------------------------------
    for (int i = tid; i < ROWS * H1PAD; i += 256) h1s[i] = (_Float16)0.0f;
    float c1[4] = {0.f, 0.f, 0.f, 0.f};     // c1 for this thread's (col, 4 rows)
    const int ecol = tid & 63;              // elementwise column 0..63
    const int erow = (tid >> 6) * 4;        // elementwise first row
    float c2 = 0.0f;                        // wave0 lanes 0..15: c2[row]
    if (tid < ROWS) {
        xbuf[tid]  = seq[0 * BSZ + batch0 + tid];   // carry init: x_in = seq[0]
        h2buf[tid] = 0.0f;
    }

    for (int t = 0; t < T_STEPS; ++t) {
        __syncthreads();   // x, h1s ready for this step

        // ---- layer 1: z = h1 @ W_hh1^T + x*W_ih1 + b1 ----------------------
        // A (16x32 MxK) layout: lane row m=ln; elem i -> K = kc*32 + (i<8?0:16) + hi*8 + (i&7)
        F16V A[2];
        #pragma unroll
        for (int kc = 0; kc < 2; ++kc) {
            const _Float16* rp = &h1s[ln * H1PAD + kc * 32 + hi * 8];
            A[kc].q[0] = *(const v4h*)(rp);
            A[kc].q[1] = *(const v4h*)(rp + 4);
            A[kc].q[2] = *(const v4h*)(rp + 16);
            A[kc].q[3] = *(const v4h*)(rp + 20);
        }
        v4f xlo = *(const v4f*)&xbuf[hi * 8];
        v4f xhi = *(const v4f*)&xbuf[hi * 8 + 4];

        #pragma unroll
        for (int j = 0; j < 2; ++j) {
            v8f acc;
            #pragma unroll
            for (int v = 0; v < 4; ++v) acc[v]     = fmaf(xlo[v], wih1c[j], b1c[j]);
            #pragma unroll
            for (int v = 0; v < 4; ++v) acc[4 + v] = fmaf(xhi[v], wih1c[j], b1c[j]);
            acc = __builtin_amdgcn_wmma_f32_16x16x32_f16(false, A[0].v, false, Bw[j][0].v,
                                                         (short)0, acc, false, false);
            acc = __builtin_amdgcn_wmma_f32_16x16x32_f16(false, A[1].v, false, Bw[j][1].v,
                                                         (short)0, acc, false, false);
            const int zc = gate * 64 + (npair * 2 + j) * 16 + ln;
            *(v4f*)&zT[zc * ZPAD + hi * 8]     = (v4f){acc[0], acc[1], acc[2], acc[3]};
            *(v4f*)&zT[zc * ZPAD + hi * 8 + 4] = (v4f){acc[4], acc[5], acc[6], acc[7]};
        }

        __syncthreads();   // all gates of z available

        // ---- layer-1 elementwise: c1/h1 update -----------------------------
        {
            v4f iz = *(const v4f*)&zT[(0 * 64 + ecol) * ZPAD + erow];
            v4f fz = *(const v4f*)&zT[(1 * 64 + ecol) * ZPAD + erow];
            v4f gz = *(const v4f*)&zT[(2 * 64 + ecol) * ZPAD + erow];
            v4f oz = *(const v4f*)&zT[(3 * 64 + ecol) * ZPAD + erow];
            #pragma unroll
            for (int r = 0; r < 4; ++r) {
                float cn = sigf(fz[r]) * c1[r] + sigf(iz[r]) * tanh_fast(gz[r]);
                float hn = sigf(oz[r]) * tanh_fast(cn);
                c1[r] = cn;
                h1s[(erow + r) * H1PAD + ecol] = (_Float16)hn;
                if (t == T_STEPS - 1) {
                    out[T_STEPS * BSZ + (batch0 + erow + r) * H1 + ecol]            = hn; // h1
                    out[T_STEPS * BSZ + BSZ * H1 + (batch0 + erow + r) * H1 + ecol] = cn; // c1
                }
            }
        }

        __syncthreads();   // h1 updated

        // ---- layer 2 (wave 0): z2 = h1 @ W_ih2^T + h2*W_hh2 + b2 -----------
        if (wave == 0) {
            F16V A2[2];
            #pragma unroll
            for (int kc = 0; kc < 2; ++kc) {
                const _Float16* rp = &h1s[ln * H1PAD + kc * 32 + hi * 8];
                A2[kc].q[0] = *(const v4h*)(rp);
                A2[kc].q[1] = *(const v4h*)(rp + 4);
                A2[kc].q[2] = *(const v4h*)(rp + 16);
                A2[kc].q[3] = *(const v4h*)(rp + 20);
            }
            v4f hlo = *(const v4f*)&h2buf[hi * 8];
            v4f hhi = *(const v4f*)&h2buf[hi * 8 + 4];
            v8f acc;
            #pragma unroll
            for (int v = 0; v < 4; ++v) acc[v]     = fmaf(hlo[v], whh2c, b2c);
            #pragma unroll
            for (int v = 0; v < 4; ++v) acc[4 + v] = fmaf(hhi[v], whh2c, b2c);
            acc = __builtin_amdgcn_wmma_f32_16x16x32_f16(false, A2[0].v, false, B2[0].v,
                                                         (short)0, acc, false, false);
            acc = __builtin_amdgcn_wmma_f32_16x16x32_f16(false, A2[1].v, false, B2[1].v,
                                                         (short)0, acc, false, false);
            if (ln < 4) {     // C layout: VGPR v, lane-half hi -> row v + 8*hi
                #pragma unroll
                for (int v = 0; v < 8; ++v)
                    z2buf[(hi * 8 + v) * 4 + ln] = acc[v];
            }
            // same wave: DS ops complete in order, backend inserts s_wait_dscnt
            if (lane < 16) {
                const int b  = batch0 + lane;
                v4f z2 = *(const v4f*)&z2buf[lane * 4];      // i,f,g,o
                float c2n = sigf(z2[1]) * c2 + sigf(z2[0]) * tanh_fast(z2[2]);
                float h2n = sigf(z2[3]) * tanh_fast(c2n);
                c2 = c2n;
                out[t * BSZ + b] = h2n;
                h2buf[lane] = h2n;
                float xn = h2n;
                if (t + 1 < T_STEPS) {
                    float s = seq[(t + 1) * BSZ + b];
                    xn = use_gt[t + 1] ? s : h2n;            // scheduled sampling
                }
                xbuf[lane] = xn;
                if (t == T_STEPS - 1) {
                    out[T_STEPS * BSZ + 2 * BSZ * H1 + b]       = h2n;  // h2
                    out[T_STEPS * BSZ + 2 * BSZ * H1 + BSZ + b] = c2n;  // c2
                }
            }
        }
    }
}

extern "C" void kernel_launch(void* const* d_in, const int* in_sizes, int n_in,
                              void* d_out, int out_size, void* d_ws, size_t ws_size,
                              hipStream_t stream) {
    (void)in_sizes; (void)n_in; (void)out_size; (void)d_ws; (void)ws_size;
    const float*         seq    = (const float*)d_in[0];
    const unsigned char* use_gt = (const unsigned char*)d_in[1];
    const float*         W_ih1  = (const float*)d_in[2];
    const float*         W_hh1  = (const float*)d_in[3];
    const float*         b_ih1  = (const float*)d_in[4];
    const float*         b_hh1  = (const float*)d_in[5];
    const float*         W_ih2  = (const float*)d_in[6];
    const float*         W_hh2  = (const float*)d_in[7];
    const float*         b_ih2  = (const float*)d_in[8];
    const float*         b_hh2  = (const float*)d_in[9];
    float*               out    = (float*)d_out;

    lstm2_scan_kernel<<<BSZ / ROWS, 256, 0, stream>>>(
        seq, use_gt, W_ih1, W_hh1, b_ih1, b_hh1, W_ih2, W_hh2, b_ih2, b_hh2, out);
}
